// PatchDiscriminator_42640435315470
// MI455X (gfx1250) — matrix operand order repaired and checked by reading, not compiled, not run on hardware
//
#include <hip/hip_runtime.h>
#include <hip/hip_bf16.h>

// Problem constants
#define Bb   16
#define Ll   4096
#define Vv   256
#define Dd   512
#define Hh   8
#define Ff   2048
#define NLl  8
#define Pp   16
#define NCc  4
#define Ss   257     // valid sequence length (256 patches + CLS)
#define SP   272     // padded sequence length (Bb*SP = 4352 = 34*128)
#define DH   64

typedef __attribute__((ext_vector_type(16))) __bf16 v16bf;
typedef __attribute__((ext_vector_type(2)))  __bf16 v2bf;
typedef __attribute__((ext_vector_type(8)))  float  v8f;

union BF16x16 {
  v16bf v;
  unsigned int u[8];
  uint4 q[2];
};

// Native f32->bf16 (RNE) pack: lowers to hardware cvt on gfx1250.
__device__ __forceinline__ unsigned int pk2(float a, float b) {
  v2bf h;
  h[0] = (__bf16)a;
  h[1] = (__bf16)b;
  return __builtin_bit_cast(unsigned int, h);
}
__device__ __forceinline__ float gelu_f(float x) {
  return 0.5f * x * (1.f + tanhf(0.7978845608f * (x + 0.044715f * x * x * x)));
}

// ---------------------------------------------------------------------------
// Weight transpose + convert: outT[n*K + k] = (bf16) in[k*N + n].  32x32 tiles.
// ---------------------------------------------------------------------------
__global__ __launch_bounds__(256) void wtrans(const float* __restrict__ in,
                                              __bf16* __restrict__ outT,
                                              int K, int N) {
  __shared__ float s[32][33];
  int n0 = blockIdx.x * 32, k0 = blockIdx.y * 32;
  int t = threadIdx.x;
  int tn = t & 31, tk = t >> 5;
#pragma unroll
  for (int j = 0; j < 4; ++j)
    s[tk + 8 * j][tn] = in[(size_t)(k0 + tk + 8 * j) * N + (n0 + tn)];
  __syncthreads();
  int wk = t & 31, wn = t >> 5;
#pragma unroll
  for (int j = 0; j < 4; ++j)
    outT[(size_t)(n0 + wn + 8 * j) * K + (k0 + wk)] = (__bf16)s[wk][wn + 8 * j];
}

// ---------------------------------------------------------------------------
// Fold embedding into conv weight, writing bf16 TRANSPOSED directly:
//   WcT[e*(Pp*Vv) + p*Vv+v] = sum_d W_emb[v,d] * W_conv[e,d,p]
// ---------------------------------------------------------------------------
__global__ __launch_bounds__(256) void wc_pre(const float* __restrict__ W_emb,
                                              const float* __restrict__ W_conv,
                                              __bf16* __restrict__ WcT) {
  int idx = blockIdx.x * 256 + threadIdx.x;     // Dd * (Pp*Vv) = 2,097,152
  int pv = idx & 4095;
  int e  = idx >> 12;
  int p = pv >> 8, v = pv & 255;
  const float* we  = W_emb + (size_t)v * Dd;
  const float* wcv = W_conv + (size_t)e * (Dd * Pp) + p;
  float acc = 0.f;
  for (int d = 0; d < Dd; ++d) acc += we[d] * wcv[d * Pp];
  WcT[(size_t)e * (Pp * Vv) + pv] = (__bf16)acc;
}

__global__ void wc_bias(const float* __restrict__ b_emb, const float* __restrict__ b_conv,
                        const float* __restrict__ W_conv, float* __restrict__ biasc) {
  int e = blockIdx.x * 256 + threadIdx.x;
  if (e >= Dd) return;
  float acc = b_conv[e];
  for (int d = 0; d < Dd; ++d) {
    float be = b_emb[d];
    const float* wp = W_conv + (size_t)e * (Dd * Pp) + d * Pp;
    for (int p = 0; p < Pp; ++p) acc += be * wp[p];
  }
  biasc[e] = acc;
}

// ---------------------------------------------------------------------------
// Generic tiled WMMA GEMM: C[M,N] = act(A[M,K](f32) * BT[N,K](bf16)^T + bias)
// Block tile 128x64, K-step 32, 256 threads = 8 waves (4x2), wave tile 32x32.
// B tiles are DMA'd into LDS with global_load_async_to_lds_b128 (ASYNCcnt),
// overlapping with the f32->bf16 staging of the A tile.
// ---------------------------------------------------------------------------
#define LDAS 40
#define LDBS 40

__global__ __launch_bounds__(256) void gemm_bf16(
    const float* __restrict__ A, int lda,
    const __bf16* __restrict__ BT,          // [N][K] row-major bf16
    const float* __restrict__ bias,
    float* __restrict__ C, int ldc,
    int K, int act) {
  __shared__ __align__(16) unsigned short sA[128 * LDAS];   // A tile row-major bf16
  __shared__ __align__(16) unsigned short sB[64 * LDBS];    // Bt[n][k] bf16
  const int tid = threadIdx.x;
  const int rowBase = blockIdx.y * 128;
  const int colBase = blockIdx.x * 64;
  const int wave = tid >> 5, lane = tid & 31;
  const int wm = wave >> 1, wn = wave & 1;
  const int lmod = lane & 15, lhalf = lane >> 4;
  const int kbA = lhalf * 8, kbB = lhalf * 16;
  // Per-thread async-copy slot for B: one b128 (8 bf16 along K) each.
  const int bn = tid >> 2;                  // 0..63
  const int bkq = (tid & 3) * 8;            // 0,8,16,24
  const __bf16* bsrc = BT + (size_t)(colBase + bn) * K + bkq;
  const unsigned bdst = (unsigned)(size_t)&sB[bn * LDBS + bkq];
  v8f acc[2][2] = {};

  for (int k0 = 0; k0 < K; k0 += 32) {
    // Kick off B-tile DMA (global -> LDS), tracked by ASYNCcnt.
    asm volatile("global_load_async_to_lds_b128 %0, %1, off"
                 :: "v"(bdst), "v"(bsrc + k0) : "memory");
    // Stage A: 128x32 f32 -> bf16 (4 float4 per thread, loads batched)
    float4 fa[4];
#pragma unroll
    for (int i = 0; i < 4; ++i) {
      int idx = tid + i * 256;
      int r = idx >> 3, c4 = idx & 7;
      fa[i] = *(const float4*)(A + (size_t)(rowBase + r) * lda + (k0 + c4 * 4));
    }
#pragma unroll
    for (int i = 0; i < 4; ++i) {
      int idx = tid + i * 256;
      int r = idx >> 3, c4 = idx & 7;
      *(uint2*)&sA[r * LDAS + c4 * 4] = make_uint2(pk2(fa[i].x, fa[i].y), pk2(fa[i].z, fa[i].w));
    }
    if (k0 + 32 < K)   // gfx1250 global_prefetch_b8 for next A tile
      __builtin_prefetch(A + (size_t)(rowBase + (tid >> 1)) * lda + (k0 + 32), 0, 1);
    asm volatile("s_wait_asynccnt 0" ::: "memory");
    __syncthreads();

    BF16x16 af[2], bfr[2];
#pragma unroll
    for (int i = 0; i < 2; ++i) {
      int m = wm * 32 + i * 16 + lmod;
      af[i].q[0] = *(const uint4*)&sA[m * LDAS + kbA];
      af[i].q[1] = *(const uint4*)&sA[m * LDAS + 16 + kbA];
    }
#pragma unroll
    for (int j = 0; j < 2; ++j) {
      int n = wn * 32 + j * 16 + lmod;
      bfr[j].q[0] = *(const uint4*)&sB[n * LDBS + kbA];
      bfr[j].q[1] = *(const uint4*)&sB[n * LDBS + 16 + kbA];
    }
#pragma unroll
    for (int i = 0; i < 2; ++i)
#pragma unroll
      for (int j = 0; j < 2; ++j)
        acc[i][j] = __builtin_amdgcn_wmma_f32_16x16x32_bf16(
            false, af[i].v, false, bfr[j].v, (short)0, acc[i][j], false, false);
    __syncthreads();
  }

#pragma unroll
  for (int i = 0; i < 2; ++i)
#pragma unroll
    for (int j = 0; j < 2; ++j) {
      int n = colBase + wn * 32 + j * 16 + lmod;
      float bv = bias ? bias[n] : 0.f;
#pragma unroll
      for (int e = 0; e < 8; ++e) {
        int m = rowBase + wm * 32 + i * 16 + lhalf * 8 + e;
        float val = acc[i][j][e] + bv;
        if (act == 1) val = gelu_f(val);
        C[(size_t)m * ldc + n] = val;
      }
    }
}

// ---------------------------------------------------------------------------
// Assemble x[B,SP,D]: CLS + pos, conv output + pos, zero padding rows
// ---------------------------------------------------------------------------
__global__ void assemble(const float* __restrict__ conv, const float* __restrict__ cls,
                         const float* __restrict__ pos, float* __restrict__ x) {
  size_t idx = (size_t)blockIdx.x * 256 + threadIdx.x;
  int d = (int)(idx & (Dd - 1));
  int s = (int)((idx >> 9) % SP);
  int b = (int)(idx / ((size_t)Dd * SP));
  float val;
  if (s == 0)        val = cls[d] + pos[d];
  else if (s < Ss)   val = conv[((size_t)b * 256 + (s - 1)) * Dd + d] + pos[(size_t)s * Dd + d];
  else               val = 0.f;
  x[idx] = val;
}

// ---------------------------------------------------------------------------
// Rotary + phi (elu+1) applied in-place to q and k. One block per (b, s<Ss).
// ---------------------------------------------------------------------------
__global__ __launch_bounds__(256) void rotary_phi(float* __restrict__ qb, float* __restrict__ kb) {
  int r = blockIdx.x;
  int b = r / Ss, s = r - b * Ss;
  int t = threadIdx.x;
  int h = t >> 5, i = t & 31;
  float ang = (float)s * __expf(-(float)i * 0.28782313663f);   // ln(10000)/32
  float sn, cs;
  __sincosf(ang, &sn, &cs);
  size_t base = ((size_t)(b * SP + s) * Hh + h) * DH;
  float* bufs[2] = {qb, kb};
#pragma unroll
  for (int w = 0; w < 2; ++w) {
    float* p = bufs[w];
    float x1 = p[base + i], x2 = p[base + 32 + i];
    float o1 = x1 * cs - x2 * sn;
    float o2 = x2 * cs + x1 * sn;
    o1 = o1 > 0.f ? o1 + 1.f : __expf(o1);
    o2 = o2 > 0.f ? o2 + 1.f : __expf(o2);
    p[base + i] = o1;
    p[base + 32 + i] = o2;
  }
}

// ---------------------------------------------------------------------------
// kv[b,h,d,m] = sum_{s<Ss} k[b,s,h,d] * v[b,s,h,m]  -- WMMA, one block per (b,h)
// ---------------------------------------------------------------------------
__global__ __launch_bounds__(128) void kv_gemm(const float* __restrict__ kbuf,
                                               const float* __restrict__ vbuf,
                                               float* __restrict__ kvout) {
  __shared__ __align__(16) unsigned short sK[64 * LDAS];   // Kt[d][s]
  __shared__ __align__(16) unsigned short sV[64 * LDAS];   // Vt[m][s]
  const int bh = blockIdx.x, b = bh >> 3, h = bh & 7;
  const int tid = threadIdx.x, wave = tid >> 5, lane = tid & 31;
  const int lmod = lane & 15, lhalf = lane >> 4;
  const int kbA = lhalf * 8, kbB = lhalf * 16;
  const float* kb = kbuf + ((size_t)b * SP * Hh + h) * DH;
  const float* vb = vbuf + ((size_t)b * SP * Hh + h) * DH;
  v8f acc[4] = {};
  const int dcol = tid & 63, shalf = tid >> 6;

  for (int s0 = 0; s0 < Ss; s0 += 32) {
    float ak[16], av[16];
#pragma unroll
    for (int j = 0; j < 16; ++j) {
      int s1 = s0 + shalf * 16 + j;
      ak[j] = (s1 < Ss) ? kb[(size_t)s1 * Dd + dcol] : 0.f;
      av[j] = (s1 < Ss) ? vb[(size_t)s1 * Dd + dcol] : 0.f;
    }
    *(uint4*)&sK[dcol * LDAS + shalf * 16] =
        make_uint4(pk2(ak[0], ak[1]), pk2(ak[2], ak[3]), pk2(ak[4], ak[5]), pk2(ak[6], ak[7]));
    *(uint4*)&sK[dcol * LDAS + shalf * 16 + 8] =
        make_uint4(pk2(ak[8], ak[9]), pk2(ak[10], ak[11]), pk2(ak[12], ak[13]), pk2(ak[14], ak[15]));
    *(uint4*)&sV[dcol * LDAS + shalf * 16] =
        make_uint4(pk2(av[0], av[1]), pk2(av[2], av[3]), pk2(av[4], av[5]), pk2(av[6], av[7]));
    *(uint4*)&sV[dcol * LDAS + shalf * 16 + 8] =
        make_uint4(pk2(av[8], av[9]), pk2(av[10], av[11]), pk2(av[12], av[13]), pk2(av[14], av[15]));
    __syncthreads();

    BF16x16 af;
    int dm = wave * 16 + lmod;
    af.q[0] = *(const uint4*)&sK[dm * LDAS + kbA];
    af.q[1] = *(const uint4*)&sK[dm * LDAS + 16 + kbA];
#pragma unroll
    for (int mt = 0; mt < 4; ++mt) {
      BF16x16 bfr;
      int n = mt * 16 + lmod;
      bfr.q[0] = *(const uint4*)&sV[n * LDAS + kbA];
      bfr.q[1] = *(const uint4*)&sV[n * LDAS + 16 + kbA];
      acc[mt] = __builtin_amdgcn_wmma_f32_16x16x32_bf16(
          false, af.v, false, bfr.v, (short)0, acc[mt], false, false);
    }
    __syncthreads();
  }
#pragma unroll
  for (int mt = 0; mt < 4; ++mt) {
    int m = mt * 16 + lmod;
#pragma unroll
    for (int e = 0; e < 8; ++e) {
      int d = wave * 16 + lhalf * 8 + e;
      kvout[((size_t)(b * Hh + h) * DH + d) * DH + m] = acc[mt][e];
    }
  }
}

// ksum[b,h,d] = sum_{s<Ss} k[b,s,h,d]
__global__ void ksum_k(const float* __restrict__ kbuf, float* __restrict__ ks) {
  int bh = blockIdx.x, b = bh >> 3, h = bh & 7;
  int d = threadIdx.x;
  const float* kb = kbuf + ((size_t)b * SP * Hh + h) * DH + d;
  float acc = 0.f;
  for (int s = 0; s < Ss; ++s) acc += kb[(size_t)s * Dd];
  ks[(size_t)bh * DH + d] = acc;
}

// z[b,s,h] = 1/(q . ksum + 1e-6)
__global__ void zcalc(const float* __restrict__ qb, const float* __restrict__ ks,
                      float* __restrict__ z) {
  int idx = blockIdx.x * 256 + threadIdx.x;
  if (idx >= Bb * Ss * Hh) return;
  int h = idx & 7;
  int s = (idx >> 3) % Ss;
  int b = idx / (Ss * Hh);
  const float* qr = qb + ((size_t)(b * SP + s) * Hh + h) * DH;
  const float* kr = ks + (size_t)(b * Hh + h) * DH;
  float acc = 0.f;
#pragma unroll 8
  for (int d = 0; d < DH; ++d) acc += qr[d] * kr[d];
  z[((size_t)(b * SP) + s) * Hh + h] = 1.f / (acc + 1e-6f);
}

// ---------------------------------------------------------------------------
// o[b,s,h,m] = z[b,s,h] * sum_d q[b,s,h,d]*kv[b,h,d,m] -- WMMA
// ---------------------------------------------------------------------------
__global__ __launch_bounds__(128) void o_gemm(const float* __restrict__ qb,
                                              const float* __restrict__ kvm,
                                              const float* __restrict__ zb,
                                              float* __restrict__ ob) {
  __shared__ __align__(16) unsigned short sKV[64 * 72];   // KVt[m][d]
  const int bh = blockIdx.x, b = bh >> 3, h = bh & 7;
  const int tid = threadIdx.x, wave = tid >> 5, lane = tid & 31;
  const int lmod = lane & 15, lhalf = lane >> 4;
  const int kbA = lhalf * 8, kbB = lhalf * 16;
  {
    int m = tid & 63, dh2 = tid >> 6;
    const float* kp = kvm + (size_t)(b * Hh + h) * DH * DH + m;
    float tv[32];
#pragma unroll
    for (int i = 0; i < 32; ++i) tv[i] = kp[(size_t)(dh2 * 32 + i) * DH];
#pragma unroll
    for (int i = 0; i < 16; ++i)
      *(unsigned int*)&sKV[m * 72 + dh2 * 32 + 2 * i] = pk2(tv[2 * i], tv[2 * i + 1]);
  }
  __syncthreads();

  int sbase = blockIdx.y * 64 + wave * 16;
  int s = sbase + lmod;
  int sc = (s < SP) ? s : 0;
  const float* ap = qb + ((size_t)(b * SP + sc) * Hh + h) * DH;
  v8f acc[4] = {};
  for (int kc = 0; kc < DH; kc += 32) {
    BF16x16 af;
    const float* a0 = ap + kc + kbA;
    float4 f0 = *(const float4*)(a0);
    float4 f1 = *(const float4*)(a0 + 4);
    float4 f2 = *(const float4*)(a0 + 16);
    float4 f3 = *(const float4*)(a0 + 20);
    af.u[0] = pk2(f0.x, f0.y); af.u[1] = pk2(f0.z, f0.w);
    af.u[2] = pk2(f1.x, f1.y); af.u[3] = pk2(f1.z, f1.w);
    af.u[4] = pk2(f2.x, f2.y); af.u[5] = pk2(f2.z, f2.w);
    af.u[6] = pk2(f3.x, f3.y); af.u[7] = pk2(f3.z, f3.w);
#pragma unroll
    for (int mt = 0; mt < 4; ++mt) {
      BF16x16 bfr;
      int n = mt * 16 + lmod;
      bfr.q[0] = *(const uint4*)&sKV[n * 72 + kc + kbA];
      bfr.q[1] = *(const uint4*)&sKV[n * 72 + kc + 16 + kbA];
      acc[mt] = __builtin_amdgcn_wmma_f32_16x16x32_bf16(
          false, af.v, false, bfr.v, (short)0, acc[mt], false, false);
    }
  }
  float zr[8];
#pragma unroll
  for (int e = 0; e < 8; ++e) {
    int sr = sbase + lhalf * 8 + e;
    zr[e] = (sr < Ss) ? zb[((size_t)(b * SP) + sr) * Hh + h] : 0.f;
  }
#pragma unroll
  for (int mt = 0; mt < 4; ++mt) {
    int m = mt * 16 + lmod;
#pragma unroll
    for (int e = 0; e < 8; ++e) {
      int sr = sbase + lhalf * 8 + e;
      if (sr < Ss)
        ob[((size_t)(b * SP + sr) * Hh + h) * DH + m] = acc[mt][e] * zr[e];
    }
  }
}

// x = LN(x + res) rowwise (res optional). One block per (b, s<Ss), 256 threads.
__global__ __launch_bounds__(256) void ln_fused(float* __restrict__ x,
                                                const float* __restrict__ res,
                                                const float* __restrict__ g,
                                                const float* __restrict__ bt) {
  __shared__ float rs[256];
  __shared__ float rq[256];
  int r = blockIdx.x;
  int b = r / Ss, s = r - b * Ss;
  float* xr = x + ((size_t)b * SP + s) * Dd;
  int t = threadIdx.x;
  float v0 = xr[t], v1 = xr[t + 256];
  if (res) {
    const float* rr = res + ((size_t)b * SP + s) * Dd;
    v0 += rr[t];
    v1 += rr[t + 256];
  }
  rs[t] = v0 + v1;
  rq[t] = v0 * v0 + v1 * v1;
  __syncthreads();
  for (int off = 128; off > 0; off >>= 1) {
    if (t < off) { rs[t] += rs[t + off]; rq[t] += rq[t + off]; }
    __syncthreads();
  }
  float mean = rs[0] * (1.f / Dd);
  float var = rq[0] * (1.f / Dd) - mean * mean;
  float rstd = rsqrtf(var + 1e-5f);
  xr[t]       = (v0 - mean) * rstd * g[t] + bt[t];
  xr[t + 256] = (v1 - mean) * rstd * g[t + 256] + bt[t + 256];
}

// out[b,c] = x[b,0,:] . W_out[:,c] + b_out[c]
__global__ void head_out(const float* __restrict__ x, const float* __restrict__ W,
                         const float* __restrict__ bo_, float* __restrict__ out) {
  int t = threadIdx.x;
  if (t >= Bb * NCc) return;
  int b = t >> 2, c = t & 3;
  const float* xr = x + (size_t)b * SP * Dd;
  float acc = bo_[c];
  for (int d = 0; d < Dd; ++d) acc += xr[d] * W[d * NCc + c];
  out[t] = acc;
}

// ---------------------------------------------------------------------------
extern "C" void kernel_launch(void* const* d_in, const int* in_sizes, int n_in,
                              void* d_out, int out_size, void* d_ws, size_t ws_size,
                              hipStream_t stream) {
  (void)in_sizes; (void)n_in; (void)out_size; (void)ws_size;
  const float* inputs  = (const float*)d_in[0];
  const float* W_emb   = (const float*)d_in[1];
  const float* b_emb   = (const float*)d_in[2];
  const float* W_conv  = (const float*)d_in[3];
  const float* b_conv  = (const float*)d_in[4];
  const float* cls_tok = (const float*)d_in[5];
  const float* pos_tab = (const float*)d_in[6];
  const float* Wq = (const float*)d_in[7];
  const float* bq = (const float*)d_in[8];
  const float* Wk = (const float*)d_in[9];
  const float* bk = (const float*)d_in[10];
  const float* Wv = (const float*)d_in[11];
  const float* bv = (const float*)d_in[12];
  const float* Wo = (const float*)d_in[13];
  const float* bo = (const float*)d_in[14];
  const float* ln1g = (const float*)d_in[15];
  const float* ln1b = (const float*)d_in[16];
  const float* ln2g = (const float*)d_in[17];
  const float* ln2b = (const float*)d_in[18];
  const float* W1 = (const float*)d_in[19];
  const float* b1 = (const float*)d_in[20];
  const float* W2 = (const float*)d_in[21];
  const float* b2 = (const float*)d_in[22];
  const float* lnfg = (const float*)d_in[23];
  const float* lnfb = (const float*)d_in[24];
  const float* Wout = (const float*)d_in[25];
  const float* bout = (const float*)d_in[26];
  float* out = (float*)d_out;

  float* ws = (float*)d_ws;
  const size_t nXD = (size_t)Bb * SP * Dd;           // 2,228,224 floats
  float* biasc = ws;
  float* x     = biasc + Dd;
  float* q     = x + nXD;
  float* k     = q + nXD;
  float* v     = k + nXD;
  float* o     = v + nXD;
  float* tmp   = o + nXD;                            // conv out / attn out / ff2 out
  float* ff1   = tmp + nXD;                          // B*SP*F
  float* kvb   = ff1 + (size_t)Bb * SP * Ff;
  float* ks    = kvb + (size_t)Bb * Hh * DH * DH;
  float* z     = ks + (size_t)Bb * Hh * DH;
  // bf16 transposed weight region
  __bf16* WcT = (__bf16*)(z + (size_t)Bb * SP * Hh);
  __bf16* WqT = WcT + (size_t)Dd * (Pp * Vv);
  __bf16* WkT = WqT + (size_t)NLl * Dd * Dd;
  __bf16* WvT = WkT + (size_t)NLl * Dd * Dd;
  __bf16* WoT = WvT + (size_t)NLl * Dd * Dd;
  __bf16* W1T = WoT + (size_t)NLl * Dd * Dd;
  __bf16* W2T = W1T + (size_t)NLl * Dd * Ff;

  // --- Weight preparation (once per launch, deterministic) ---
  wc_pre<<<(Dd * Pp * Vv) / 256, 256, 0, stream>>>(W_emb, W_conv, WcT);
  wc_bias<<<2, 256, 0, stream>>>(b_emb, b_conv, W_conv, biasc);
  const dim3 tDD(Dd / 32, Dd / 32);     // K=512, N=512
  for (int l = 0; l < NLl; ++l) {
    wtrans<<<tDD, 256, 0, stream>>>(Wq + (size_t)l * Dd * Dd, WqT + (size_t)l * Dd * Dd, Dd, Dd);
    wtrans<<<tDD, 256, 0, stream>>>(Wk + (size_t)l * Dd * Dd, WkT + (size_t)l * Dd * Dd, Dd, Dd);
    wtrans<<<tDD, 256, 0, stream>>>(Wv + (size_t)l * Dd * Dd, WvT + (size_t)l * Dd * Dd, Dd, Dd);
    wtrans<<<tDD, 256, 0, stream>>>(Wo + (size_t)l * Dd * Dd, WoT + (size_t)l * Dd * Dd, Dd, Dd);
    wtrans<<<dim3(Ff / 32, Dd / 32), 256, 0, stream>>>(
        W1 + (size_t)l * Dd * Ff, W1T + (size_t)l * Dd * Ff, Dd, Ff);
    wtrans<<<dim3(Dd / 32, Ff / 32), 256, 0, stream>>>(
        W2 + (size_t)l * Ff * Dd, W2T + (size_t)l * Ff * Dd, Ff, Dd);
  }

  // --- Fused patch-embed GEMM (M=4096, K=4096, N=512) ---
  gemm_bf16<<<dim3(Dd / 64, (Bb * 256) / 128), 256, 0, stream>>>(
      inputs, Pp * Vv, WcT, biasc, tmp, Dd, Pp * Vv, 0);
  assemble<<<(unsigned)(nXD / 256), 256, 0, stream>>>(tmp, cls_tok, pos_tab, x);

  const dim3 gD(Dd / 64, (Bb * SP) / 128);   // (8, 34)
  const dim3 gF(Ff / 64, (Bb * SP) / 128);   // (32, 34)
  for (int l = 0; l < NLl; ++l) {
    gemm_bf16<<<gD, 256, 0, stream>>>(x, Dd, WqT + (size_t)l * Dd * Dd,
                                      bq + (size_t)l * Dd, q, Dd, Dd, 0);
    gemm_bf16<<<gD, 256, 0, stream>>>(x, Dd, WkT + (size_t)l * Dd * Dd,
                                      bk + (size_t)l * Dd, k, Dd, Dd, 0);
    gemm_bf16<<<gD, 256, 0, stream>>>(x, Dd, WvT + (size_t)l * Dd * Dd,
                                      bv + (size_t)l * Dd, v, Dd, Dd, 0);
    rotary_phi<<<Bb * Ss, 256, 0, stream>>>(q, k);
    kv_gemm<<<Bb * Hh, 128, 0, stream>>>(k, v, kvb);
    ksum_k<<<Bb * Hh, 64, 0, stream>>>(k, ks);
    zcalc<<<(Bb * Ss * Hh + 255) / 256, 256, 0, stream>>>(q, ks, z);
    o_gemm<<<dim3(Bb * Hh, 5), 128, 0, stream>>>(q, kvb, z, o);
    gemm_bf16<<<gD, 256, 0, stream>>>(o, Dd, WoT + (size_t)l * Dd * Dd,
                                      bo + (size_t)l * Dd, tmp, Dd, Dd, 0);
    ln_fused<<<Bb * Ss, 256, 0, stream>>>(x, tmp, ln1g + (size_t)l * Dd, ln1b + (size_t)l * Dd);
    gemm_bf16<<<gF, 256, 0, stream>>>(x, Dd, W1T + (size_t)l * Dd * Ff,
                                      b1 + (size_t)l * Ff, ff1, Ff, Dd, 1);
    gemm_bf16<<<gD, 256, 0, stream>>>(ff1, Ff, W2T + (size_t)l * Ff * Dd,
                                      b2 + (size_t)l * Dd, tmp, Dd, Ff, 0);
    ln_fused<<<Bb * Ss, 256, 0, stream>>>(x, tmp, ln2g + (size_t)l * Dd, ln2b + (size_t)l * Dd);
  }
  ln_fused<<<Bb * Ss, 256, 0, stream>>>(x, nullptr, lnfg, lnfb);
  head_out<<<1, 64, 0, stream>>>(x, Wout, bout, out);
}
